// LinearLayer_4114578670344
// MI455X (gfx1250) — matrix operand order repaired and checked by reading, not compiled, not run on hardware
//
#include <hip/hip_runtime.h>
#include <hip/hip_bf16.h>
#include <math.h>

typedef __attribute__((ext_vector_type(16))) __bf16 v16bf;
typedef __attribute__((ext_vector_type(8)))  __bf16 v8bf;
typedef __attribute__((ext_vector_type(8)))  float  v8f;

// ---------------------------------------------------------------------------
// Repack fp32 weight W[K][V] (row major) into bf16 WMMA B-fragment order,
// folding in the e3nn path-normalization scale (1/sqrt(K)).
// Packed layout: P[kstep][ntile][lane][e], 512 bf16 per (kstep,ntile) fragment.
// Per ISA 7.12.2 (16-bit B 32x16): lane half h, column = lane&15,
// element e -> K = 8*h + e (e<8), 16 + 8*h + (e-8) (e>=8) within the 32-wide k-step.
// ---------------------------------------------------------------------------
__global__ __launch_bounds__(256)
void pack_weights(const float* __restrict__ W, __bf16* __restrict__ P,
                  int K, int V, float scale) {
  int idx = blockIdx.x * blockDim.x + threadIdx.x;
  if (idx >= K * V) return;
  int e     = idx & 15;
  int lane  = (idx >> 4) & 31;
  int frag  = idx >> 9;
  int ntiles = V >> 4;
  int ntile = frag % ntiles;
  int kstep = frag / ntiles;
  int half  = lane >> 4;
  int k = kstep * 32 + half * 8 + e + (e >= 8 ? 8 : 0);
  int n = ntile * 16 + (lane & 15);
  P[idx] = (__bf16)(W[k * V + n] * scale);
}

// ---------------------------------------------------------------------------
// Fused per-irrep-block pipeline:  X[32,U,D] -> H[32,V,D] -> gate -> Y[32,U,D]
// One workgroup (8 wave32s) per 32 rows of N; each wave computes two 16x16
// output tiles per column tile so every B fragment (from L2) feeds 2 WMMAs.
// Hidden tensor lives entirely in LDS. U = mul_in, V = mul_hid, D = 2l+1.
// ---------------------------------------------------------------------------
template<int U, int V, int D, int OFF>
__global__ __launch_bounds__(256)
void fused_block(const float* __restrict__ x,
                 const __bf16* __restrict__ W1p,
                 const __bf16* __restrict__ W2p,
                 float* __restrict__ out) {
  constexpr int LD = 960;   // full feature row stride (floats)
  constexpr int NW = 8;     // waves per workgroup
  constexpr int TM = 32;    // rows of N per workgroup (two 16-row M tiles)

  extern __shared__ char smem[];
  __bf16* Xbf = (__bf16*)smem;                                        // [TM][D][U]
  float*  H   = (float*) (smem + TM*D*U*2);                           // [TM][D][V]
  __bf16* A2  = (__bf16*)(smem + TM*D*U*2 + TM*D*V*4);                // [TM][D][V]
  float*  Ob  = (float*) (smem + TM*D*U*2 + TM*D*V*4 + TM*D*V*2);     // [TM][U*D]

  const int tid  = threadIdx.x;
  const int n0   = blockIdx.x * TM;
  // Force wave id into an SGPR: tile loops become scalar branches and EXEC
  // is provably all-1s around every WMMA (ISA 7.12 requirement).
  const int wv   = __builtin_amdgcn_readfirstlane(tid >> 5);
  const int lane = tid & 31;
  const int half = lane >> 4;
  const int c16  = lane & 15;

  // ---- Phase 0: load X tile (coalesced), convert to bf16, u-contiguous in LDS
  for (int i = tid; i < TM * D * U; i += 256) {
    int m = i / (U * D), c = i % (U * D);
    int u = c / D, dd = c % D;
    Xbf[(m * D + dd) * U + u] = (__bf16)x[(size_t)(n0 + m) * LD + OFF + c];
  }
  __syncthreads();

  // ---- Phase 1: GEMM1  H[m,dd,v] = sum_u Xbf[m,dd,u] * W1[u,v]/sqrt(U)
  for (int t = wv; t < D * (V / 16); t += NW) {
    int dd = t / (V / 16), nt = t % (V / 16);
    v8f acc0 = {}, acc1 = {};
    for (int ks = 0; ks < U / 32; ++ks) {
      const __bf16* ar0 = &Xbf[( c16       * D + dd) * U + ks * 32 + 8 * half];
      const __bf16* ar1 = &Xbf[((c16 + 16) * D + dd) * U + ks * 32 + 8 * half];
      v8bf a0l = *(const v8bf*)ar0,        a1l = *(const v8bf*)ar1;
      v8bf a0h = *(const v8bf*)(ar0 + 16), a1h = *(const v8bf*)(ar1 + 16);
      v16bf a0 = __builtin_shufflevector(a0l, a0h, 0,1,2,3,4,5,6,7,8,9,10,11,12,13,14,15);
      v16bf a1 = __builtin_shufflevector(a1l, a1h, 0,1,2,3,4,5,6,7,8,9,10,11,12,13,14,15);
      v16bf b  = *(const v16bf*)(W1p + ((size_t)(ks * (V / 16) + nt) * 32 + lane) * 16);
      acc0 = __builtin_amdgcn_wmma_f32_16x16x32_bf16(false, a0, false, b,
                                                     (short)0, acc0, false, false);
      acc1 = __builtin_amdgcn_wmma_f32_16x16x32_bf16(false, a1, false, b,
                                                     (short)0, acc1, false, false);
    }
    #pragma unroll
    for (int r = 0; r < 8; ++r) {
      H[(((r + 8 * half     ) * D + dd) * V) + nt * 16 + c16] = acc0[r];
      H[(((r + 8 * half + 16) * D + dd) * V) + nt * 16 + c16] = acc1[r];
    }
  }
  __syncthreads();

  // ---- Phase 2: norm-activation: gelu(||h||)/||h|| = 0.5*(1+erf(||h||/sqrt2))
  for (int i = tid; i < TM * V; i += 256) {
    int m = i / V, vc = i % V;
    float h[D];
    float s = 0.f;
    #pragma unroll
    for (int dd = 0; dd < D; ++dd) {
      h[dd] = H[(m * D + dd) * V + vc];
      s += h[dd] * h[dd];
    }
    float scale = 0.5f * (1.0f + erff(sqrtf(s) * 0.70710678118654752f));
    #pragma unroll
    for (int dd = 0; dd < D; ++dd)
      A2[(m * D + dd) * V + vc] = (__bf16)(h[dd] * scale);
  }
  __syncthreads();

  // ---- Phase 3: GEMM2  Y[m,dd,w] = sum_v A2[m,dd,v] * W2[v,w]/sqrt(V)
  for (int t = wv; t < D * (U / 16); t += NW) {
    int dd = t / (U / 16), nt = t % (U / 16);
    v8f acc0 = {}, acc1 = {};
    for (int ks = 0; ks < V / 32; ++ks) {
      const __bf16* ar0 = &A2[( c16       * D + dd) * V + ks * 32 + 8 * half];
      const __bf16* ar1 = &A2[((c16 + 16) * D + dd) * V + ks * 32 + 8 * half];
      v8bf a0l = *(const v8bf*)ar0,        a1l = *(const v8bf*)ar1;
      v8bf a0h = *(const v8bf*)(ar0 + 16), a1h = *(const v8bf*)(ar1 + 16);
      v16bf a0 = __builtin_shufflevector(a0l, a0h, 0,1,2,3,4,5,6,7,8,9,10,11,12,13,14,15);
      v16bf a1 = __builtin_shufflevector(a1l, a1h, 0,1,2,3,4,5,6,7,8,9,10,11,12,13,14,15);
      v16bf b  = *(const v16bf*)(W2p + ((size_t)(ks * (U / 16) + nt) * 32 + lane) * 16);
      acc0 = __builtin_amdgcn_wmma_f32_16x16x32_bf16(false, a0, false, b,
                                                     (short)0, acc0, false, false);
      acc1 = __builtin_amdgcn_wmma_f32_16x16x32_bf16(false, a1, false, b,
                                                     (short)0, acc1, false, false);
    }
    #pragma unroll
    for (int r = 0; r < 8; ++r) {
      Ob[(r + 8 * half     ) * (U * D) + (nt * 16 + c16) * D + dd] = acc0[r];
      Ob[(r + 8 * half + 16) * (U * D) + (nt * 16 + c16) * D + dd] = acc1[r];
    }
  }
  __syncthreads();

  // ---- Phase 4: coalesced fp32 store of this block's output columns
  for (int i = tid; i < TM * U * D; i += 256) {
    int m = i / (U * D), c = i % (U * D);
    out[(size_t)(n0 + m) * LD + OFF + c] = Ob[i];
  }
}

// ---------------------------------------------------------------------------
extern "C" void kernel_launch(void* const* d_in, const int* in_sizes, int n_in,
                              void* d_out, int out_size, void* d_ws, size_t ws_size,
                              hipStream_t stream) {
  (void)in_sizes; (void)n_in; (void)out_size; (void)ws_size;
  const float* feat = (const float*)d_in[0];
  const float* W1_0 = (const float*)d_in[1];
  const float* W1_1 = (const float*)d_in[2];
  const float* W1_2 = (const float*)d_in[3];
  const float* W2_0 = (const float*)d_in[4];
  const float* W2_1 = (const float*)d_in[5];
  const float* W2_2 = (const float*)d_in[6];
  float* out = (float*)d_out;

  // packed bf16 weights in workspace (~0.7 MB total; L2 resident)
  __bf16* P10 = (__bf16*)d_ws;
  __bf16* P11 = P10 + 256 * 512;
  __bf16* P12 = P11 + 128 * 256;
  __bf16* P20 = P12 + 64 * 128;
  __bf16* P21 = P20 + 512 * 256;
  __bf16* P22 = P21 + 256 * 128;

  auto pack = [&](const float* W, __bf16* P, int K, int V) {
    int total = K * V;
    pack_weights<<<(total + 255) / 256, 256, 0, stream>>>(
        W, P, K, V, 1.0f / sqrtf((float)K));
  };
  pack(W1_0, P10, 256, 512);
  pack(W1_1, P11, 128, 256);
  pack(W1_2, P12,  64, 128);
  pack(W2_0, P20, 512, 256);
  pack(W2_1, P21, 256, 128);
  pack(W2_2, P22, 128,  64);

  const int N = 32768;
  const int grid = N / 32;
  // dynamic LDS: 192*D*(U+V) bytes  (144 / 216 / 180 KB)
  fused_block<256, 512, 1,   0><<<grid, 256, 192 * 1 * (256 + 512), stream>>>(feat, P10, P20, out);
  fused_block<128, 256, 3, 256><<<grid, 256, 192 * 3 * (128 + 256), stream>>>(feat, P11, P21, out);
  fused_block< 64, 128, 5, 640><<<grid, 256, 192 * 5 * ( 64 + 128), stream>>>(feat, P12, P22, out);
}
